// ParallelDecoderLayer_37563783970970
// MI455X (gfx1250) — matrix operand order repaired and checked by reading, not compiled, not run on hardware
//
#include <hip/hip_runtime.h>
#include <stdint.h>

// Problem constants (match reference)
#define NB   128        // batch
#define NS   512        // encoder positions
#define ND   1024       // d_model
#define NH   8
#define NDK  64
#define NM   40
#define ROWS (NB * NS)  // 65536

typedef __attribute__((ext_vector_type(16))) __bf16 bf16x16;
typedef __attribute__((ext_vector_type(4)))  __bf16 bf16x4;
typedef __attribute__((ext_vector_type(8)))  float  f32x8;

struct __align__(16) Frag { uint4 a, b; };

static __device__ __forceinline__ __bf16 f2bf(float f) {
  union { float f; uint32_t u; } v; v.f = f;
  uint32_t r = v.u + 0x7fffu + ((v.u >> 16) & 1u);
  uint16_t h = (uint16_t)(r >> 16);
  return __builtin_bit_cast(__bf16, h);
}
static __device__ __forceinline__ float bf2f(__bf16 x) {
  union { uint32_t u; float f; } v;
  v.u = ((uint32_t)__builtin_bit_cast(uint16_t, x)) << 16;
  return v.f;
}
static __device__ __forceinline__ float sigm(float x) { return 1.0f / (1.0f + __expf(-x)); }

// ---- CDNA5 async global->LDS copy (ASYNCcnt path, CDNA5 ISA) ----
static __device__ __forceinline__ void async_ld_b64(unsigned lds_off, const void* g) {
  asm volatile("global_load_async_to_lds_b64 %0, %1, off" :: "v"(lds_off), "v"(g) : "memory");
}
static __device__ __forceinline__ void wait_async0() {
  asm volatile("s_wait_asynccnt 0" ::: "memory");
}

// ---- CDNA5 Tensor Data Mover staging (TENSORcnt path, CDNA5 ISA/8) ----
#if __has_builtin(__builtin_amdgcn_tensor_load_to_lds)
#define HAVE_TDM 1
typedef unsigned __attribute__((ext_vector_type(4))) v4u;
typedef int      __attribute__((ext_vector_type(4))) v4i;
typedef int      __attribute__((ext_vector_type(8))) v8i;
// Load a [rows x 32] tile of 2-byte elements (row stride 1024 elems) into LDS.
static __device__ __forceinline__ void tdm_load_tile(unsigned lds_off, const void* gaddr, int rows) {
  uint64_t ga = (uint64_t)gaddr;
  v4u g0;
  g0[0] = 1u;                                                // count=1 valid descriptor
  g0[1] = lds_off;                                           // lds_addr (bytes)
  g0[2] = (unsigned)ga;                                      // global_addr[31:0]
  g0[3] = (unsigned)((ga >> 32) & 0x1FFFFFFu) | (2u << 30);  // global_addr[56:32] | type=2 (image)
  v8i g1;
  g1[0] = (1 << 16);          // workgroup_mask=0 | data_size=1 (2 bytes)
  g1[1] = (32 << 16);         // atomic_barrier_addr=0 | tensor_dim0[15:0] = 32
  g1[2] = (rows << 16);       // tensor_dim0[31:16]=0 | tensor_dim1[15:0] = rows
  g1[3] = (32 << 16);         // tensor_dim1[31:16]=0 | tile_dim0 = 32
  g1[4] = rows;               // tile_dim1 = rows | tile_dim2 = 0
  g1[5] = 1024;               // tensor_dim0_stride[31:0] = 1024 elems (row pitch)
  g1[6] = 0;                  // tensor_dim0_stride[47:32] | tensor_dim1_stride lo
  g1[7] = 0;                  // tensor_dim1_stride hi
  v4i zz = {0, 0, 0, 0};
#if __has_include(<hip/amd_detail/amd_gfx1250_TDM.h>)
  v8i z8 = {0, 0, 0, 0, 0, 0, 0, 0};
  __builtin_amdgcn_tensor_load_to_lds(g0, g1, zz, zz, z8, 0);  // 6-arg toolchain
#else
  __builtin_amdgcn_tensor_load_to_lds(g0, g1, zz, zz, 0);      // 5-arg ROCm 7.2 toolchain
#endif
}
#else
#define HAVE_TDM 0
#endif

// ---- WMMA fragment loaders per ISA 7.12.2 (wave32) ----
// A (16x32 bf16): lane<16 holds row M=lane, K in {0..7,16..23}; lane>=16: K {8..15,24..31}
static __device__ __forceinline__ bf16x16 ldfragA(const __bf16* tile, int lane) {
  const uint4* p = (const uint4*)(tile + (lane & 15) * 32);
  int c = lane >> 4;
  Frag f; f.a = p[c]; f.b = p[c + 2];
  return __builtin_bit_cast(bf16x16, f);
}
// B (32x16 bf16) from Bt stored [N][K]: lane<16 holds col N=lane, K in {0..7,16..23}; lane>=16: K {8..15,24..31}
static __device__ __forceinline__ bf16x16 ldfragB(const __bf16* btRow0, int lane, int ldK) {
  const uint4* p = (const uint4*)(btRow0 + (size_t)(lane & 15) * ldK);
  int c = lane >> 4;
  Frag f; f.a = p[c]; f.b = p[c + 2];
  return __builtin_bit_cast(bf16x16, f);
}

// =====================================================================
// Big WMMA GEMM over enc_output (bf16).  A:[ROWS][1024]  Bt:[1024][1024] (n-major)
// Block = 512 threads (16 waves), tile 64 rows x full N=1024 (64 cols/wave),
// K-loop: 32 steps of 32; A tile double-buffered in LDS.
// STAGE 0: per-lane async global->LDS (ASYNCcnt); STAGE 1: TDM descriptor load (TENSORcnt).
// MODE 0: scores[row] = sum_n tanh(acc + hp[b][n]) * wav[n]   (additive attention)
// MODE 1: Cbf[row][n] = bf16(acc + bias[n])                   (K|V projection)
// =====================================================================
template <int MODE, int STAGE>
__global__ void __launch_bounds__(512)
wmma_gemm(const __bf16* __restrict__ A, const __bf16* __restrict__ Bt,
          const float* __restrict__ e0, const float* __restrict__ e1,
          float* __restrict__ scores, __bf16* __restrict__ Cbf) {
  __shared__ __align__(16) __bf16 Atile[2][64 * 32];  // 2 x 4KB
  __shared__ float red[64];
  const int tid  = threadIdx.x;
  const int wave = tid >> 5, lane = tid & 31;
  const int half = lane >> 4, nIdx = lane & 15;
  const int row0 = blockIdx.x * 64;

  f32x8 acc[4][4];
  const f32x8 z8 = {0.f, 0.f, 0.f, 0.f, 0.f, 0.f, 0.f, 0.f};
#pragma unroll
  for (int i = 0; i < 4; ++i)
#pragma unroll
    for (int j = 0; j < 4; ++j) acc[i][j] = z8;

  auto stage = [&](int buf, int ks) {
#if HAVE_TDM
    if (STAGE == 1) {
      if (tid == 0)
        tdm_load_tile((unsigned)(size_t)&Atile[buf][0],
                      (const void*)(A + (size_t)row0 * ND + ks * 32), 64);
      return;
    }
#endif
    int r = tid >> 3, c = tid & 7;  // 64 rows x 8 chunks of 8B
    unsigned lds = (unsigned)(size_t)&Atile[buf][r * 32 + c * 4];
    const void* g = (const void*)(A + (size_t)(row0 + r) * ND + ks * 32 + c * 4);
    async_ld_b64(lds, g);
  };
  auto wait_stage = [&]() {
#if HAVE_TDM
    if (STAGE == 1) { __builtin_amdgcn_s_wait_tensorcnt(0); return; }
#endif
    wait_async0();
  };

  stage(0, 0);
  for (int ks = 0; ks < 32; ++ks) {
    const int cur = ks & 1;
    wait_stage();
    __syncthreads();                           // staged A tile visible to all waves
    if (ks + 1 < 32) stage(cur ^ 1, ks + 1);   // prefetch next K-slice
    bf16x16 af[4];
#pragma unroll
    for (int mt = 0; mt < 4; ++mt) af[mt] = ldfragA(&Atile[cur][mt * 16 * 32], lane);
    const __bf16* bbase = Bt + (size_t)(wave * 64) * ND + ks * 32;
#pragma unroll
    for (int nt = 0; nt < 4; ++nt) {
      bf16x16 bf = ldfragB(bbase + (size_t)nt * 16 * ND, lane, ND);
#pragma unroll
      for (int mt = 0; mt < 4; ++mt)
        acc[mt][nt] = __builtin_amdgcn_wmma_f32_16x16x32_bf16(false, af[mt], false, bf,
                                                              (short)0, acc[mt][nt], false, false);
    }
    __syncthreads();                           // everyone done with Atile[cur] before overwrite
  }

  if (MODE == 0) {
    if (tid < 64) red[tid] = 0.f;
    __syncthreads();
    const int b = row0 >> 9;                   // 512 rows per batch element, 64 | 512
    float part[4][8];
#pragma unroll
    for (int mt = 0; mt < 4; ++mt)
#pragma unroll
      for (int r = 0; r < 8; ++r) part[mt][r] = 0.f;
#pragma unroll
    for (int nt = 0; nt < 4; ++nt) {
      int n = wave * 64 + nt * 16 + nIdx;
      float hpv = e0[b * ND + n];
      float wv  = e1[n];
#pragma unroll
      for (int mt = 0; mt < 4; ++mt)
#pragma unroll
        for (int r = 0; r < 8; ++r)
          part[mt][r] += tanhf(acc[mt][nt][r] + hpv) * wv;
    }
#pragma unroll
    for (int mt = 0; mt < 4; ++mt)
#pragma unroll
      for (int r = 0; r < 8; ++r) {
        float v = part[mt][r];
        v += __shfl_xor(v, 1); v += __shfl_xor(v, 2);
        v += __shfl_xor(v, 4); v += __shfl_xor(v, 8);
        if (nIdx == 0) atomicAdd(&red[mt * 16 + half * 8 + r], v);  // ds_add_f32
      }
    __syncthreads();
    if (tid < 64) scores[row0 + tid] = red[tid];
  } else {
#pragma unroll
    for (int nt = 0; nt < 4; ++nt) {
      int n = wave * 64 + nt * 16 + nIdx;
      float bv = e0[n];
#pragma unroll
      for (int mt = 0; mt < 4; ++mt)
#pragma unroll
        for (int r = 0; r < 8; ++r) {
          int m = row0 + mt * 16 + half * 8 + r;
          Cbf[(size_t)m * ND + n] = f2bf(acc[mt][nt][r] + bv);
        }
    }
  }
}

// ---------------- fp32 -> bf16 bulk convert (vectorized) ----------------
__global__ void conv_bf16_v4(const float4* __restrict__ in, bf16x4* __restrict__ out, int n4) {
  int i = blockIdx.x * blockDim.x + threadIdx.x;
  if (i < n4) {
    float4 v = in[i];
    bf16x4 o; o.x = f2bf(v.x); o.y = f2bf(v.y); o.z = f2bf(v.z); o.w = f2bf(v.w);
    out[i] = o;
  }
}

// W[K][N] fp32 -> Wt[N][K] bf16
__global__ void transpose_bf16(const float* __restrict__ W, __bf16* __restrict__ Wt, int K, int N) {
  __shared__ float t[32][33];
  int kb = blockIdx.x * 32, nb = blockIdx.y * 32;
  for (int i = threadIdx.y; i < 32; i += 8)
    t[i][threadIdx.x] = W[(size_t)(kb + i) * N + nb + threadIdx.x];
  __syncthreads();
  for (int i = threadIdx.y; i < 32; i += 8)
    Wt[(size_t)(nb + i) * K + kb + threadIdx.x] = f2bf(t[threadIdx.x][i]);
}

// ---------------- small fp32 GEMM: C[M][N] (+)= act(A[M][K] @ W + bias) ----------------
// WT: W stored [N][K] (torch Linear weight);  !WT: W stored [K][N].
template <bool WT, int ACT, bool ACCUM>
__global__ void __launch_bounds__(128)
gemm_f32(const float* __restrict__ A, const float* __restrict__ W,
         const float* __restrict__ bias, float* __restrict__ C, int M, int N, int K) {
  __shared__ float Arow[4][2048];
  const int m0 = blockIdx.y * 4;
  const int n  = blockIdx.x * 128 + threadIdx.x;
  for (int mi = 0; mi < 4; ++mi)
    for (int k = threadIdx.x; k < K; k += 128)
      Arow[mi][k] = A[(size_t)(m0 + mi) * K + k];
  __syncthreads();
  if (n >= N) return;
  float acc[4];
  float b0 = bias ? bias[n] : 0.f;
  for (int mi = 0; mi < 4; ++mi) acc[mi] = b0;
  if (WT) {
    const float* wr = W + (size_t)n * K;
    for (int k = 0; k < K; ++k) {
      float w = wr[k];
#pragma unroll
      for (int mi = 0; mi < 4; ++mi) acc[mi] += Arow[mi][k] * w;
    }
  } else {
    for (int k = 0; k < K; ++k) {
      float w = W[(size_t)k * N + n];
#pragma unroll
      for (int mi = 0; mi < 4; ++mi) acc[mi] += Arow[mi][k] * w;
    }
  }
#pragma unroll
  for (int mi = 0; mi < 4; ++mi) {
    float v = acc[mi];
    if (ACT == 1) v = fmaxf(v, 0.f);
    size_t idx = (size_t)(m0 + mi) * N + n;
    C[idx] = ACCUM ? (C[idx] + v) : v;
  }
}

// ---------------- LSTM pointwise ----------------
__global__ void lstm_pw(const float* __restrict__ g, const float* __restrict__ cprev,
                        float* __restrict__ h, float* __restrict__ c,
                        float* __restrict__ hExtra, int extraStride) {
  int b = blockIdx.y, d = blockIdx.x * 256 + threadIdx.x;
  const float* gb = g + (size_t)b * 4096;
  float i  = sigm(gb[d]);
  float f  = sigm(gb[1024 + d]);
  float gg = tanhf(gb[2048 + d]);
  float o  = sigm(gb[3072 + d]);
  float c2 = f * cprev[b * ND + d] + i * gg;
  float hv = o * tanhf(c2);
  c[b * ND + d] = c2;
  h[b * ND + d] = hv;
  if (hExtra) hExtra[(size_t)b * extraStride + d] = hv;
}

// ---------------- codebook softmax (rows of 100) ----------------
__global__ void softmax_cb(float* __restrict__ p) {
  __shared__ float sh[128];
  int b = blockIdx.x, t = threadIdx.x;
  float v = (t < 100) ? p[b * 100 + t] : -3.4e38f;
  sh[t] = v; __syncthreads();
  for (int s = 64; s > 0; s >>= 1) { if (t < s) sh[t] = fmaxf(sh[t], sh[t + s]); __syncthreads(); }
  float mx = sh[0]; __syncthreads();
  float e = (t < 100) ? __expf(v - mx) : 0.f;
  sh[t] = e; __syncthreads();
  for (int s = 64; s > 0; s >>= 1) { if (t < s) sh[t] += sh[t + s]; __syncthreads(); }
  if (t < 100) p[b * 100 + t] = e / sh[0];
}

// ---------------- masked softmax over S=512 ----------------
__global__ void __launch_bounds__(512)
softmax_alpha(const float* __restrict__ sc, const uint8_t* __restrict__ mask,
              float* __restrict__ alpha) {
  __shared__ float sh[512];
  int b = blockIdx.x, t = threadIdx.x;
  float v = mask[b * NS + t] ? -1e9f : sc[b * NS + t];
  sh[t] = v; __syncthreads();
  for (int s = 256; s > 0; s >>= 1) { if (t < s) sh[t] = fmaxf(sh[t], sh[t + s]); __syncthreads(); }
  float mx = sh[0]; __syncthreads();
  float e = __expf(v - mx);
  sh[t] = e; __syncthreads();
  for (int s = 256; s > 0; s >>= 1) { if (t < s) sh[t] += sh[t + s]; __syncthreads(); }
  alpha[b * NS + t] = e / sh[0];
}

// ---------------- v = alpha @ enc (bf16 enc, L2-resident) ----------------
__global__ void weighted_sum(const float* __restrict__ alpha, const __bf16* __restrict__ enc16,
                             float* __restrict__ v) {
  __shared__ float al[512];
  int b = blockIdx.y, d = blockIdx.x * 256 + threadIdx.x;
  for (int s = threadIdx.x; s < NS; s += 256) al[s] = alpha[b * NS + s];
  __syncthreads();
  float acc = 0.f;
  const __bf16* eb = enc16 + (size_t)b * NS * ND + d;
  for (int s = 0; s < NS; ++s) acc += al[s] * bf2f(eb[(size_t)s * ND]);
  v[b * ND + d] = acc;
}

// ---------------- meshed-memory attention (per b,h), kv bf16 [ROWS][1024] ----------------
__global__ void __launch_bounds__(256)
mha_attend(const float* __restrict__ q, const __bf16* __restrict__ kv,
           const float* __restrict__ mk, const float* __restrict__ mv,
           const uint8_t* __restrict__ mask, float* __restrict__ out) {
  __shared__ float sc[NS + NM];
  __shared__ float qh[NDK];
  __shared__ float red[256];
  int h = blockIdx.x, b = blockIdx.y, t = threadIdx.x;
  if (t < NDK) qh[t] = q[(size_t)b * 512 + h * NDK + t];
  __syncthreads();
  const float sK = 8.0f;                // sqrt(DK)
  const float sM = 6.324555320336759f;  // sqrt(M=40)
  for (int j = t; j < NS + NM; j += 256) {
    float s = 0.f;
    if (j < NS) {
      if (mask[b * NS + j]) { sc[j] = -1e9f; continue; }
      const __bf16* kp = kv + ((size_t)(b * NS + j)) * 1024 + h * NDK;
      for (int d = 0; d < NDK; ++d) s += qh[d] * bf2f(kp[d]);
    } else {
      const float* kp = mk + (size_t)(j - NS) * 512 + h * NDK;
      for (int d = 0; d < NDK; ++d) s += qh[d] * kp[d] * sK;
    }
    sc[j] = s * 0.125f;                 // 1/sqrt(DK)
  }
  __syncthreads();
  float mx = -3.4e38f;
  for (int j = t; j < NS + NM; j += 256) mx = fmaxf(mx, sc[j]);
  red[t] = mx; __syncthreads();
  for (int s = 128; s > 0; s >>= 1) { if (t < s) red[t] = fmaxf(red[t], red[t + s]); __syncthreads(); }
  mx = red[0]; __syncthreads();
  float sum = 0.f;
  for (int j = t; j < NS + NM; j += 256) { float e = __expf(sc[j] - mx); sc[j] = e; sum += e; }
  red[t] = sum; __syncthreads();
  for (int s = 128; s > 0; s >>= 1) { if (t < s) red[t] += red[t + s]; __syncthreads(); }
  float inv = 1.f / red[0];
  if (t < NDK) {
    float acc = 0.f;
    for (int j = 0; j < NS; ++j)
      acc += sc[j] * bf2f(kv[((size_t)(b * NS + j)) * 1024 + 512 + h * NDK + t]);
    for (int j = 0; j < NM; ++j)
      acc += sc[NS + j] * mv[(size_t)j * 512 + h * NDK + t] * sM;
    out[(size_t)b * 512 + h * NDK + t] = acc * inv;
  }
}

// ---------------- small elementwise glue ----------------
__global__ void build_x1(const float* __restrict__ nw, const float* __restrict__ em,
                         const float* __restrict__ sh, float* __restrict__ x1) {
  int b = blockIdx.y, d = blockIdx.x * 256 + threadIdx.x;
  const float* s1 = sh + NB * ND;
  const float* s2 = sh + 2 * NB * ND;
  x1[(size_t)b * 2048 + d]        = (em[b * ND + d] + s1[b * ND + d] + s2[b * ND + d]) * (1.f / 3.f);
  x1[(size_t)b * 2048 + 1024 + d] = nw[b * ND + d];
}
__global__ void add_relu(const float* __restrict__ pre, const float* __restrict__ h1,
                         float* __restrict__ h1n) {
  int i = blockIdx.x * 256 + threadIdx.x;
  h1n[i] = fmaxf(pre[i], 0.f) + h1[i];
}
__global__ void concat2(const float* __restrict__ a, const float* __restrict__ b_,
                        float* __restrict__ o) {
  int b = blockIdx.y, d = blockIdx.x * 256 + threadIdx.x;
  o[(size_t)b * 2048 + d]        = a[b * ND + d];
  o[(size_t)b * 2048 + 1024 + d] = b_[b * ND + d];
}
__global__ void build_fv(const float* __restrict__ mha, const float* __restrict__ h1n,
                         const float* __restrict__ vc, float* __restrict__ fv) {
  int b = blockIdx.y, d = blockIdx.x * 256 + threadIdx.x;
  fv[(size_t)b * 2048 + d]        = mha[b * ND + d] + h1n[b * ND + d];
  fv[(size_t)b * 2048 + 1024 + d] = vc[b * ND + d];
}
__global__ void gate_out(const float* __restrict__ gbr, const float* __restrict__ ibr,
                         float* __restrict__ h3, float* __restrict__ outRegion) {
  int b = blockIdx.y, d = blockIdx.x * 256 + threadIdx.x;
  float v = sigm(gbr[b * ND + d]) * ibr[b * ND + d];
  h3[b * ND + d] = v;
  outRegion[(size_t)b * 2048 + 1024 + d] = v;
}
__global__ void concat_bias(const float* __restrict__ bk, const float* __restrict__ bv,
                            float* __restrict__ bkv) {
  int i = blockIdx.x * 256 + threadIdx.x;
  if (i < 512) bkv[i] = bk[i];
  else if (i < 1024) bkv[i] = bv[i - 512];
}

// =====================================================================
extern "C" void kernel_launch(void* const* d_in, const int* in_sizes, int n_in,
                              void* d_out, int out_size, void* d_ws, size_t ws_size,
                              hipStream_t stream) {
  (void)in_sizes; (void)n_in; (void)out_size; (void)ws_size;
  const float* next_word = (const float*)d_in[0];
  const float* enc_mean  = (const float*)d_in[1];
  const float* enc       = (const float*)d_in[2];
  const float* state_h   = (const float*)d_in[3];
  const float* state_c   = (const float*)d_in[4];
  const uint8_t* sg_mask = (const uint8_t*)d_in[5];
  const float* W_ih1 = (const float*)d_in[6],  *W_hh1 = (const float*)d_in[7];
  const float* b_ih1 = (const float*)d_in[8],  *b_hh1 = (const float*)d_in[9];
  const float* W_ih2 = (const float*)d_in[10], *W_hh2 = (const float*)d_in[11];
  const float* b_ih2 = (const float*)d_in[12], *b_hh2 = (const float*)d_in[13];
  const float* param = (const float*)d_in[14];
  const float* Wa_e  = (const float*)d_in[15], *Wa_h = (const float*)d_in[16];
  const float* wa_v  = (const float*)d_in[17];
  const float* Wq = (const float*)d_in[18], *bq = (const float*)d_in[19];
  const float* Wk = (const float*)d_in[20], *bk = (const float*)d_in[21];
  const float* Wv = (const float*)d_in[22], *bv = (const float*)d_in[23];
  const float* Wo = (const float*)d_in[24], *bo = (const float*)d_in[25];
  const float* m_k = (const float*)d_in[26], *m_v = (const float*)d_in[27];
  const float* Wl  = (const float*)d_in[28], *bl  = (const float*)d_in[29];
  const float* Wl1 = (const float*)d_in[30], *bl1 = (const float*)d_in[31];
  const float* Wl2 = (const float*)d_in[32], *bl2 = (const float*)d_in[33];
  const float* Wl3 = (const float*)d_in[34], *bl3 = (const float*)d_in[35];

  // ---- workspace carve-up ----
  char* ws = (char*)d_ws;
  size_t off = 0;
  auto alloc = [&](size_t bytes) { size_t o = off; off = (off + bytes + 255) & ~(size_t)255; return o; };
  __bf16* enc16 = (__bf16*)(ws + alloc((size_t)ROWS * ND * 2));   // 134 MB (fits MI455X 192MB L2)
  __bf16* waeT  = (__bf16*)(ws + alloc((size_t)ND * ND * 2));
  __bf16* wkvT  = (__bf16*)(ws + alloc((size_t)ND * ND * 2));     // rows 0-511 Wk^T, 512-1023 Wv^T
  __bf16* kv16  = (__bf16*)(ws + alloc((size_t)ROWS * ND * 2));   // cols 0-511 K, 512-1023 V
  float* x1     = (float*)(ws + alloc((size_t)NB * 2048 * 4));
  float* gates  = (float*)(ws + alloc((size_t)NB * 4096 * 4));
  float* cb     = (float*)(ws + alloc((size_t)NB * 128 * 4));
  float* h1npre = (float*)(ws + alloc((size_t)NB * ND * 4));
  float* h1n    = (float*)(ws + alloc((size_t)NB * ND * 4));
  float* hp     = (float*)(ws + alloc((size_t)NB * ND * 4));
  float* scores = (float*)(ws + alloc((size_t)ROWS * 4));
  float* alpha  = (float*)(ws + alloc((size_t)ROWS * 4));
  float* vctx   = (float*)(ws + alloc((size_t)NB * ND * 4));
  float* x2     = (float*)(ws + alloc((size_t)NB * 2048 * 4));
  float* qbuf   = (float*)(ws + alloc((size_t)NB * 512 * 4));
  float* mhah   = (float*)(ws + alloc((size_t)NB * 512 * 4));
  float* mhao   = (float*)(ws + alloc((size_t)NB * ND * 4));
  float* fv     = (float*)(ws + alloc((size_t)NB * 2048 * 4));
  float* ibr    = (float*)(ws + alloc((size_t)NB * ND * 4));
  float* gbr    = (float*)(ws + alloc((size_t)NB * ND * 4));
  float* bkv    = (float*)(ws + alloc((size_t)1024 * 4));

  // ---- output carve-up: [out(B,2D)][h(3,B,D)][c(2,B,D)] ----
  float* outf = (float*)d_out;
  float* h1p = outf + NB * 2048;
  float* h2p = h1p + NB * ND;
  float* h3p = h2p + NB * ND;
  float* c1p = h3p + NB * ND;
  float* c2p = c1p + NB * ND;

  const dim3 bd256(256);
  const dim3 gBD(4, NB);  // 4*256 = 1024 = D

  // Phase 0: precision prep (enc -> bf16, weights -> [N][K] bf16, bias concat)
  conv_bf16_v4<<<(ROWS * ND / 4 + 255) / 256, bd256, 0, stream>>>((const float4*)enc, (bf16x4*)enc16, ROWS * ND / 4);
  transpose_bf16<<<dim3(32, 32), dim3(32, 8), 0, stream>>>(Wa_e, waeT, ND, ND);
  transpose_bf16<<<dim3(32, 16), dim3(32, 8), 0, stream>>>(Wk, wkvT, ND, 512);
  transpose_bf16<<<dim3(32, 16), dim3(32, 8), 0, stream>>>(Wv, wkvT + (size_t)512 * ND, ND, 512);
  concat_bias<<<4, bd256, 0, stream>>>(bk, bv, bkv);

  // Phase 1: LSTM cell 1
  build_x1<<<gBD, bd256, 0, stream>>>(next_word, enc_mean, state_h, x1);
  gemm_f32<true, 0, false><<<dim3(32, 32), 128, 0, stream>>>(x1, W_ih1, b_ih1, gates, NB, 4096, 2048);
  gemm_f32<true, 0, true ><<<dim3(32, 32), 128, 0, stream>>>(state_h, W_hh1, b_hh1, gates, NB, 4096, 1024);
  lstm_pw<<<gBD, bd256, 0, stream>>>(gates, state_c, h1p, c1p, (float*)nullptr, 0);

  // Phase 2: soft codebook + residual relu
  gemm_f32<false, 0, false><<<dim3(1, 32), 128, 0, stream>>>(h1p, param, nullptr, cb, NB, 100, 1024);
  softmax_cb<<<NB, 128, 0, stream>>>(cb);
  gemm_f32<true, 0, false><<<dim3(8, 32), 128, 0, stream>>>(cb, param, nullptr, h1npre, NB, 1024, 100);
  add_relu<<<NB * ND / 256, bd256, 0, stream>>>(h1npre, h1p, h1n);

  // Phase 3: additive attention (big WMMA GEMM 1, async-LDS staging, fused tanh-dot epilogue)
  gemm_f32<false, 0, false><<<dim3(8, 32), 128, 0, stream>>>(h1n, Wa_h, nullptr, hp, NB, 1024, 1024);
  wmma_gemm<0, 0><<<ROWS / 64, 512, 0, stream>>>(enc16, waeT, hp, wa_v, scores, nullptr);
  softmax_alpha<<<NB, 512, 0, stream>>>(scores, sg_mask, alpha);
  weighted_sum<<<gBD, bd256, 0, stream>>>(alpha, enc16, vctx);

  // Phase 4: K/V projection (big WMMA GEMM 2, TDM staging, bf16 out, enc16 re-read from L2)
  wmma_gemm<1, 1><<<ROWS / 64, 512, 0, stream>>>(enc16, wkvT, bkv, nullptr, nullptr, kv16);

  // Phase 5: LSTM cell 2 (h2 duplicated into out[:, :D])
  concat2<<<gBD, bd256, 0, stream>>>(h1n, vctx, x2);
  gemm_f32<true, 0, false><<<dim3(32, 32), 128, 0, stream>>>(x2, W_ih2, b_ih2, gates, NB, 4096, 2048);
  gemm_f32<true, 0, true ><<<dim3(32, 32), 128, 0, stream>>>(state_h + NB * ND, W_hh2, b_hh2, gates, NB, 4096, 1024);
  lstm_pw<<<gBD, bd256, 0, stream>>>(gates, state_c + NB * ND, h2p, c2p, outf, 2048);

  // Phase 6: meshed-memory MHA
  gemm_f32<false, 0, false><<<dim3(4, 32), 128, 0, stream>>>(h1n, Wq, bq, qbuf, NB, 512, 1024);
  mha_attend<<<dim3(NH, NB), bd256, 0, stream>>>(qbuf, kv16, m_k, m_v, sg_mask, mhah);
  gemm_f32<false, 0, false><<<dim3(8, 32), 128, 0, stream>>>(mhah, Wo, bo, mhao, NB, 1024, 512);

  // Phase 7: gated dual-branch fusion (h3 into out[:, D:2D])
  build_fv<<<gBD, bd256, 0, stream>>>(mhao, h1n, vctx, fv);
  gemm_f32<false, 1, false><<<dim3(8, 32), 128, 0, stream>>>(fv, Wl,  bl,  ibr, NB, 1024, 2048);
  gemm_f32<false, 1, true ><<<dim3(8, 32), 128, 0, stream>>>(hp, Wl2, bl2, ibr, NB, 1024, 1024);
  gemm_f32<false, 1, false><<<dim3(8, 32), 128, 0, stream>>>(fv, Wl1, bl1, gbr, NB, 1024, 2048);
  gemm_f32<false, 1, true ><<<dim3(8, 32), 128, 0, stream>>>(hp, Wl3, bl3, gbr, NB, 1024, 1024);
  gate_out<<<gBD, bd256, 0, stream>>>(gbr, ibr, h3p, outf);
}
